// BiMamba_88665304858804
// MI455X (gfx1250) — matrix operand order repaired and checked by reading, not compiled, not run on hardware
//
#include <hip/hip_runtime.h>
#include <hip/hip_bf16.h>

// Problem constants (match reference)
#define B_    4
#define L_    2048
#define D_    256
#define NL_   2
#define DS_   16
#define DC_   4
#define DI_   512
#define DTR_  16
#define ROWS  (B_ * L_)        // 8192
#define XDW   (DTR_ + 2 * DS_) // 48

typedef __attribute__((ext_vector_type(16))) __bf16          v16bf;
typedef __attribute__((ext_vector_type(8)))  __bf16          v8bf;
typedef __attribute__((ext_vector_type(8)))  float           v8f;
typedef int v4i __attribute__((vector_size(16)));            // matches builtin param

#define AS1 __attribute__((address_space(1)))
#define AS3 __attribute__((address_space(3)))

#if __has_builtin(__builtin_amdgcn_global_load_async_to_lds_b128) && \
    __has_builtin(__builtin_amdgcn_s_wait_asynccnt)
#define USE_ASYNC_LDS 1
#else
#define USE_ASYNC_LDS 0
#endif

__device__ __forceinline__ unsigned short f2bf(float f) {
    // round-to-nearest-even fp32 -> bf16
    unsigned int u = __float_as_uint(f);
    u += 0x7FFFu + ((u >> 16) & 1u);
    return (unsigned short)(u >> 16);
}

// ---------------------------------------------------------------------------
// fp32 -> bf16 conversion (amortized once per operand, outside the GEMM loop)
// n must be a multiple of 4.
// ---------------------------------------------------------------------------
__global__ void cvt_bf16_k(const float* __restrict__ in,
                           unsigned short* __restrict__ out, int n)
{
    int i = (blockIdx.x * blockDim.x + threadIdx.x) * 4;
    if (i >= n) return;
    float4 v = *(const float4*)(in + i);
    ushort4 o;
    o.x = f2bf(v.x); o.y = f2bf(v.y); o.z = f2bf(v.z); o.w = f2bf(v.w);
    *(ushort4*)(out + i) = o;
}

// ---------------------------------------------------------------------------
// C[M,N] = A[M,K] * W[N,K]^T, bf16 WMMA, f32 accumulate.
// Each wave computes a 16 x (16*NT) strip: one A fragment feeds NT WMMAs.
// All fragment loads for a K-step are issued before the WMMA chain so the
// scheduler can use partial s_wait_loadcnt and overlap loads with the
// matrix pipe.
// Fragment gathers use contiguous k (same order for A and B):
//   A (16x32): lane<16 -> K 0..15;             lane>=16 -> K 16..31
//   B (32x16): lane<16 -> K 0..15;             lane>=16 -> K 16..31
// ---------------------------------------------------------------------------
template<int NT>
__global__ void __launch_bounds__(128)
wmma_gemm_bf16(const unsigned short* __restrict__ A, int lda,
               const unsigned short* __restrict__ W, int K,   // W: N x K row-major
               float* __restrict__ C, int ldc, int M, int N)
{
    const int wave = blockIdx.x * (blockDim.x >> 5) + (threadIdx.x >> 5);
    const int lane = threadIdx.x & 31;
    const int ngrp = N / (16 * NT);
    const int total = (M >> 4) * ngrp;
    if (wave >= total) return;            // wave-uniform: EXEC stays all-ones

    const int tm     = wave / ngrp;
    const int tg     = wave % ngrp;
    const int lanelo = lane & 15;
    const int hi     = lane >> 4;
    const int kbA    = hi * 16;
    const int kbB    = hi * 16;

    const unsigned short* __restrict__ arow  = A + (size_t)(tm * 16 + lanelo) * lda;
    const unsigned short* __restrict__ wbase = W + (size_t)(tg * 16 * NT + lanelo) * K;

    v8f acc[NT];
    const v8f vzero = {};
#pragma unroll
    for (int t = 0; t < NT; ++t) acc[t] = vzero;

#pragma unroll 2
    for (int k0 = 0; k0 < K; k0 += 32) {
        __builtin_prefetch(arow + k0 + 128, 0, 1);

        // Issue every load for this K-step first (one big clause).
        v8bf a_lo = *(const v8bf*)(arow + k0 + kbA);
        v8bf a_hi = *(const v8bf*)(arow + k0 + kbA + 8);
        v8bf blo[NT], bhi[NT];
#pragma unroll
        for (int t = 0; t < NT; ++t) {
            const unsigned short* wr = wbase + (size_t)(t * 16) * K;
            blo[t] = *(const v8bf*)(wr + k0 + kbB);
            bhi[t] = *(const v8bf*)(wr + k0 + kbB + 8);
        }

        v16bf av = __builtin_shufflevector(a_lo, a_hi,
                       0, 1, 2, 3, 4, 5, 6, 7, 8, 9, 10, 11, 12, 13, 14, 15);
#pragma unroll
        for (int t = 0; t < NT; ++t) {
            v16bf bv = __builtin_shufflevector(blo[t], bhi[t],
                           0, 1, 2, 3, 4, 5, 6, 7, 8, 9, 10, 11, 12, 13, 14, 15);
            acc[t] = __builtin_amdgcn_wmma_f32_16x16x32_bf16(
                false, av, false, bv, (short)0, acc[t], false, false);
        }
    }

    // D layout: lane<16 -> N=lane, M=r ; lane>=16 -> N=lane-16, M=8+r
    const int mbase = tm * 16 + hi * 8;
#pragma unroll
    for (int t = 0; t < NT; ++t) {
        const int n = tg * 16 * NT + t * 16 + lanelo;
#pragma unroll
        for (int r = 0; r < 8; ++r)
            C[(size_t)(mbase + r) * ldc + n] = acc[t][r];
    }
}

// ---------------------------------------------------------------------------
// Causal depthwise conv1d (kernel DC) + bias + silu:  xz[:, :DI] -> uc
// ---------------------------------------------------------------------------
__global__ void conv_silu_k(const float* __restrict__ xz,
                            const float* __restrict__ cw,   // DI x DC
                            const float* __restrict__ cb,   // DI
                            float* __restrict__ uc)
{
    int idx = blockIdx.x * blockDim.x + threadIdx.x;
    if (idx >= ROWS * DI_) return;
    int d   = idx % DI_;
    int row = idx / DI_;
    int l   = row % L_;
    float acc = cb[d];
#pragma unroll
    for (int k = 0; k < DC_; ++k) {
        int lk = l - (DC_ - 1) + k;
        if (lk >= 0)
            acc += cw[d * DC_ + k] * xz[(size_t)(row - (DC_ - 1) + k) * (2 * DI_) + d];
    }
    uc[idx] = acc / (1.f + __expf(-acc));
}

// ---------------------------------------------------------------------------
// delta = softplus(dt @ dtw^T + dtb),  dt = xdbl[:, :DTR]
// ---------------------------------------------------------------------------
__global__ void dt_softplus_k(const float* __restrict__ xdbl,
                              const float* __restrict__ dtw,  // DI x DTR
                              const float* __restrict__ dtb,  // DI
                              float* __restrict__ delta)
{
    int idx = blockIdx.x * blockDim.x + threadIdx.x;
    if (idx >= ROWS * DI_) return;
    int d   = idx % DI_;
    int row = idx / DI_;
    const float* dt = xdbl + (size_t)row * XDW;
    float acc = dtb[d];
#pragma unroll
    for (int r = 0; r < DTR_; ++r) acc += dt[r] * dtw[d * DTR_ + r];
    delta[idx] = (acc > 20.f) ? acc : log1pf(__expf(acc));
}

// ---------------------------------------------------------------------------
// Selective scan + gating.  One channel per thread, h[DS] in registers.
// Per-step B_t/C_t (32 floats) staged to LDS via async global->LDS b128
// copies (gfx1250 async path) when available.
// ucy holds uc on entry; gated y on exit (in-place).
// ---------------------------------------------------------------------------
__global__ void __launch_bounds__(256)
scan_k(const float* __restrict__ xz,     // z-gate at col DI_+d
       const float* __restrict__ xdbl,   // B_t, C_t at cols [16..48)
       const float* __restrict__ delta,
       const float* __restrict__ Alog,   // DI x DS
       const float* __restrict__ Dp,     // DI
       float* __restrict__ ucy)
{
    const int b    = blockIdx.x >> 1;
    const int half = blockIdx.x & 1;
    const int d    = half * 256 + threadIdx.x;

    __shared__ __align__(16) float sBC[2 * DS_];   // [0..15]=B_t, [16..31]=C_t

    float A[DS_], h[DS_];
#pragma unroll
    for (int s = 0; s < DS_; ++s) { A[s] = -__expf(Alog[d * DS_ + s]); h[s] = 0.f; }
    const float Dd = Dp[d];

    for (int l = 0; l < L_; ++l) {
        const int row = b * L_ + l;
        __syncthreads();                     // WAR on sBC
#if USE_ASYNC_LDS
        if (threadIdx.x < 8) {
            const float* g = xdbl + (size_t)row * XDW + DTR_ + threadIdx.x * 4;
            AS1 v4i* gsrc = (AS1 v4i*)((AS1 void*)(void*)g);
            AS3 v4i* ldst = (AS3 v4i*)((AS3 void*)(void*)&sBC[threadIdx.x * 4]);
            __builtin_amdgcn_global_load_async_to_lds_b128(gsrc, ldst, 0, 0);
        }
        __builtin_amdgcn_s_wait_asynccnt(0);  // wave-uniform; non-issuing waves have 0
#else
        if (threadIdx.x < 32)
            sBC[threadIdx.x] = xdbl[(size_t)row * XDW + DTR_ + threadIdx.x];
#endif
        __syncthreads();                     // RAW on sBC

        const float dl = delta[(size_t)row * DI_ + d];
        const float u  = ucy[(size_t)row * DI_ + d];
        float y = 0.f;
#pragma unroll
        for (int s = 0; s < DS_; ++s) {
            float dA = __expf(dl * A[s]);
            h[s] = dA * h[s] + dl * sBC[s] * u;
            y += h[s] * sBC[DS_ + s];
        }
        y += Dd * u;
        float z = xz[(size_t)row * (2 * DI_) + DI_ + d];
        float g = z / (1.f + __expf(-z));
        ucy[(size_t)row * DI_ + d] = y * g;
    }
}

// ---------------------------------------------------------------------------
// LayerNorm over D=256, one row per block
// ---------------------------------------------------------------------------
__global__ void __launch_bounds__(256)
layernorm_k(const float* __restrict__ in, const float* __restrict__ w,
            const float* __restrict__ bb, float* __restrict__ out)
{
    const int row = blockIdx.x;
    const int t   = threadIdx.x;
    __shared__ float red[256];

    float x = in[(size_t)row * D_ + t];
    red[t] = x; __syncthreads();
#pragma unroll
    for (int off = 128; off; off >>= 1) { if (t < off) red[t] += red[t + off]; __syncthreads(); }
    float mu = red[0] * (1.f / D_);
    __syncthreads();

    float c = x - mu;
    red[t] = c * c; __syncthreads();
#pragma unroll
    for (int off = 128; off; off >>= 1) { if (t < off) red[t] += red[t + off]; __syncthreads(); }
    float inv = rsqrtf(red[0] * (1.f / D_) + 1e-5f);

    out[(size_t)row * D_ + t] = c * inv * w[t] + bb[t];
}

// ---------------------------------------------------------------------------
// Flip along L (per batch)
// ---------------------------------------------------------------------------
__global__ void flip_k(const float* __restrict__ in, float* __restrict__ out)
{
    int idx = blockIdx.x * blockDim.x + threadIdx.x;
    if (idx >= ROWS * D_) return;
    int c   = idx % D_;
    int row = idx / D_;
    int b = row / L_, l = row % L_;
    out[idx] = in[(size_t)(b * L_ + (L_ - 1 - l)) * D_ + c];
}

// ---------------------------------------------------------------------------
// out[b,l,:] = concat(hfwd[b,l,:], hbwd[b,L-1-l,:])
// ---------------------------------------------------------------------------
__global__ void concat_k(const float* __restrict__ hf, const float* __restrict__ hb,
                         float* __restrict__ out)
{
    int idx = blockIdx.x * blockDim.x + threadIdx.x;
    if (idx >= ROWS * 2 * D_) return;
    int c   = idx % (2 * D_);
    int row = idx / (2 * D_);
    int b = row / L_, l = row % L_;
    float v;
    if (c < D_) v = hf[(size_t)row * D_ + c];
    else        v = hb[(size_t)(b * L_ + (L_ - 1 - l)) * D_ + (c - D_)];
    out[idx] = v;
}

// ---------------------------------------------------------------------------
extern "C" void kernel_launch(void* const* d_in, const int* in_sizes, int n_in,
                              void* d_out, int out_size, void* d_ws, size_t ws_size,
                              hipStream_t stream)
{
    const float* x          = (const float*)d_in[0];
    const float* in_proj_w  = (const float*)d_in[1];
    const float* conv_w     = (const float*)d_in[2];
    const float* conv_b     = (const float*)d_in[3];
    const float* x_proj_w   = (const float*)d_in[4];
    const float* dt_w       = (const float*)d_in[5];
    const float* dt_b       = (const float*)d_in[6];
    const float* A_log      = (const float*)d_in[7];
    const float* Dp         = (const float*)d_in[8];
    const float* out_proj_w = (const float*)d_in[9];
    const float* norm_w     = (const float*)d_in[10];
    const float* norm_b     = (const float*)d_in[11];
    float*       out        = (float*)d_out;

    // workspace carve-up (256B-aligned chunks)
    char*  wsb = (char*)d_ws;
    size_t off = 0;
    auto alloc = [&](size_t bytes) {
        char* p = wsb + off;
        off += (bytes + 255) & ~(size_t)255;
        return (void*)p;
    };
    float* xflip = (float*)alloc((size_t)ROWS * D_ * 4);
    float* actA  = (float*)alloc((size_t)ROWS * D_ * 4);
    float* actB  = (float*)alloc((size_t)ROWS * D_ * 4);
    float* hfwd  = (float*)alloc((size_t)ROWS * D_ * 4);
    float* xz    = (float*)alloc((size_t)ROWS * 2 * DI_ * 4);
    float* uc    = (float*)alloc((size_t)ROWS * DI_ * 4);   // gated y in-place
    float* xdbl  = (float*)alloc((size_t)ROWS * XDW * 4);
    float* delta = (float*)alloc((size_t)ROWS * DI_ * 4);
    float* tmp   = (float*)alloc((size_t)ROWS * D_ * 4);
    unsigned short* act_bf = (unsigned short*)alloc((size_t)ROWS * D_ * 2);
    unsigned short* ucy_bf = (unsigned short*)alloc((size_t)ROWS * DI_ * 2);
    unsigned short* w_bf   = (unsigned short*)alloc((size_t)2 * DI_ * D_ * 2);
    (void)ws_size; (void)in_sizes; (void)n_in; (void)out_size;

    const int ew = 256;
    auto cvt = [&](const float* src, unsigned short* dst, int n) {
        cvt_bf16_k<<<(n / 4 + ew - 1) / ew, ew, 0, stream>>>(src, dst, n);
    };

    const float* hbwd = nullptr;

    for (int dir = 0; dir < 2; ++dir) {
        const float* act;
        if (dir == 0) act = x;
        else {
            flip_k<<<(ROWS * D_ + ew - 1) / ew, ew, 0, stream>>>(x, xflip);
            act = xflip;
        }
        for (int layer = 0; layer < NL_; ++layer) {
            const int li = dir * NL_ + layer;
            const float* Win = in_proj_w  + (size_t)li * 2 * DI_ * D_;
            const float* Wc  = conv_w     + (size_t)li * DI_ * DC_;
            const float* bc  = conv_b     + (size_t)li * DI_;
            const float* Wx  = x_proj_w   + (size_t)li * XDW * DI_;
            const float* Wdt = dt_w       + (size_t)li * DI_ * DTR_;
            const float* bdt = dt_b       + (size_t)li * DI_;
            const float* Al  = A_log      + (size_t)li * DI_ * DS_;
            const float* Dpl = Dp         + (size_t)li * DI_;
            const float* Wo  = out_proj_w + (size_t)li * D_ * DI_;
            const float* nw  = norm_w     + (size_t)li * D_;
            const float* nb  = norm_b     + (size_t)li * D_;

            // ---- in_proj: (8192x256) x (1024x256)^T -> xz (8192x1024)
            cvt(act, act_bf, ROWS * D_);
            cvt(Win, w_bf, 2 * DI_ * D_);
            {
                int waves = (ROWS / 16) * (2 * DI_ / 64);   // 8192 waves (NT=4)
                wmma_gemm_bf16<4><<<(waves + 3) / 4, 128, 0, stream>>>(
                    act_bf, D_, w_bf, D_, xz, 2 * DI_, ROWS, 2 * DI_);
            }
            conv_silu_k<<<(ROWS * DI_ + ew - 1) / ew, ew, 0, stream>>>(xz, Wc, bc, uc);

            // ---- x_proj: (8192x512) x (48x512)^T -> xdbl (8192x48)
            cvt(uc, ucy_bf, ROWS * DI_);
            cvt(Wx, w_bf, XDW * DI_);
            {
                int waves = (ROWS / 16) * (XDW / 48);       // 512 waves (NT=3)
                wmma_gemm_bf16<3><<<(waves + 3) / 4, 128, 0, stream>>>(
                    ucy_bf, DI_, w_bf, DI_, xdbl, XDW, ROWS, XDW);
            }
            dt_softplus_k<<<(ROWS * DI_ + ew - 1) / ew, ew, 0, stream>>>(xdbl, Wdt, bdt, delta);

            scan_k<<<2 * B_, 256, 0, stream>>>(xz, xdbl, delta, Al, Dpl, uc);

            // ---- out_proj: (8192x512) x (256x512)^T -> tmp (8192x256)
            cvt(uc, ucy_bf, ROWS * DI_);
            cvt(Wo, w_bf, D_ * DI_);
            {
                int waves = (ROWS / 16) * (D_ / 64);        // 2048 waves (NT=4)
                wmma_gemm_bf16<4><<<(waves + 3) / 4, 128, 0, stream>>>(
                    ucy_bf, DI_, w_bf, DI_, tmp, D_, ROWS, D_);
            }

            float* nxt;
            if (dir == 0) nxt = (layer == NL_ - 1) ? hfwd : actA;
            else          nxt = (layer == NL_ - 1) ? actB : actA;
            layernorm_k<<<ROWS, 256, 0, stream>>>(tmp, nw, nb, nxt);
            act = nxt;
            if (dir == 1 && layer == NL_ - 1) hbwd = nxt;
        }
    }

    concat_k<<<(ROWS * 2 * D_ + ew - 1) / ew, ew, 0, stream>>>(hfwd, hbwd, out);
}
